// CentroidDistance_59124519796877
// MI455X (gfx1250) — matrix operand order repaired and checked by reading, not compiled, not run on hardware
//
#include <hip/hip_runtime.h>
#include <hip/hip_bf16.h>

// ---------------------------------------------------------------------------
// CentroidDistance: graph/node centroid distances via Gram identity.
//   xc = node @ cent^T   (bf16 WMMA, f32 accumulate)
//   d2 = x2 + c2 - 2*xc ; dist = sqrt(max(d2, 1e-12))
//   node_out[n,k] = dist * mask[n]
//   graph[k]      = sum_n node_out[n,k] / sum(mask)
// N=32768, K=2048, D=512 (fixed by the reference).
// ---------------------------------------------------------------------------

#define N_DIM 32768
#define K_DIM 2048
#define D_DIM 512

#define BM 128
#define BN 128
#define KSTEP 32
#define LDS_STRIDE 40 // 32 bf16 + 8 pad -> 80B rows, 16B-aligned chunks

typedef __attribute__((ext_vector_type(16))) __bf16 v16bf;
typedef __attribute__((ext_vector_type(8)))  __bf16 v8bf;
typedef __attribute__((ext_vector_type(4)))  __bf16 v4bf;
typedef __attribute__((ext_vector_type(8)))  float  v8f;

// ---------------------------------------------------------------------------
__global__ void cd_zero_acc(float* acc) {
    int i = blockIdx.x * blockDim.x + threadIdx.x;
    if (i < K_DIM) acc[i] = 0.0f;
}

// One wave per row: fp32 squared-norm, shfl tree reduce (wave32).
__global__ void cd_rownorm(const float* __restrict__ src, float* __restrict__ dst,
                           int rows) {
    int wave = threadIdx.x >> 5;
    int lane = threadIdx.x & 31;
    int r = blockIdx.x * 8 + wave;
    if (r >= rows) return;
    const float4* p = (const float4*)(src + (size_t)r * D_DIM);
    float s = 0.0f;
#pragma unroll
    for (int j = 0; j < D_DIM / (32 * 4); ++j) {
        float4 v = p[lane + 32 * j];
        s += v.x * v.x + v.y * v.y + v.z * v.z + v.w * v.w;
    }
#pragma unroll
    for (int off = 16; off > 0; off >>= 1) s += __shfl_xor(s, off);
    if (lane == 0) dst[r] = s;
}

// ---------------------------------------------------------------------------
// Tiled WMMA GEMM + fused epilogue.
// Block: 256 threads = 8 waves; block tile BM x BN = 128 x 128.
// Wave grid 2(M) x 4(N); each wave: 64 x 32 = 4 x 2 wmma 16x16 tiles.
// ---------------------------------------------------------------------------
__global__ void __launch_bounds__(256)
cd_gemm(const float* __restrict__ node, const float* __restrict__ cent,
        const float* __restrict__ mask, const float* __restrict__ x2,
        const float* __restrict__ c2, float* __restrict__ accg,
        float* __restrict__ nodeOut) {
    __shared__ __bf16 As[BM][LDS_STRIDE];
    __shared__ __bf16 Bs[BN][LDS_STRIDE];
    __shared__ float  x2s[BM];
    __shared__ float  msks[BM];
    __shared__ float  c2s[BN];

    const int tid   = threadIdx.x;
    const int lane  = tid & 31;
    const int wave  = tid >> 5;
    const int wm    = wave & 1;   // 0..1  -> 64-row slab
    const int wn    = wave >> 1;  // 0..3  -> 32-col slab
    const int row16 = lane & 15;
    const int half  = lane >> 4;

    const int gmBase = blockIdx.x * BM; // node rows
    const int gnBase = blockIdx.y * BN; // centroid cols

    // Stage per-block norm/mask vectors into LDS (covered by in-loop barriers).
    if (tid < BM) {
        x2s[tid]  = x2[gmBase + tid];
        msks[tid] = mask[gmBase + tid];
    } else {
        c2s[tid - BM] = c2[gnBase + (tid - BM)];
    }

    v8f acc[4][2] = {};

    for (int kk = 0; kk < D_DIM; kk += KSTEP) {
        // ---- stage A/B K-slab to LDS as bf16 (256 threads x 4 float4 tasks each)
#pragma unroll
        for (int j = 0; j < 4; ++j) {
            int t   = tid + 256 * j;     // 0..1023
            int row = t >> 3;            // 0..127
            int c4  = t & 7;             // 0..7 -> 4-float chunk
            float4 av = *(const float4*)(node + (size_t)(gmBase + row) * D_DIM + kk + 4 * c4);
            float4 bv = *(const float4*)(cent + (size_t)(gnBase + row) * D_DIM + kk + 4 * c4);
            v4bf a4 = { (__bf16)av.x, (__bf16)av.y, (__bf16)av.z, (__bf16)av.w };
            v4bf b4 = { (__bf16)bv.x, (__bf16)bv.y, (__bf16)bv.z, (__bf16)bv.w };
            *(v4bf*)(&As[row][4 * c4]) = a4;
            *(v4bf*)(&Bs[row][4 * c4]) = b4;
            if (kk + KSTEP < D_DIM) { // pull next slab toward the caches
                __builtin_prefetch(node + (size_t)(gmBase + row) * D_DIM + kk + KSTEP + 4 * c4, 0, 1);
                __builtin_prefetch(cent + (size_t)(gnBase + row) * D_DIM + kk + KSTEP + 4 * c4, 0, 1);
            }
        }
        __syncthreads();

        // ---- load fragments (ISA 16-bit A layout: two b128 chunks per lane)
        v16bf afr[4], bfr[2];
#pragma unroll
        for (int mi = 0; mi < 4; ++mi) {
            const __bf16* ap = &As[wm * 64 + mi * 16 + row16][0];
            v8bf lo = *(const v8bf*)(ap + half * 8);
            v8bf hi = *(const v8bf*)(ap + 16 + half * 8);
            afr[mi] = __builtin_shufflevector(lo, hi, 0, 1, 2, 3, 4, 5, 6, 7,
                                              8, 9, 10, 11, 12, 13, 14, 15);
        }
#pragma unroll
        for (int ni = 0; ni < 2; ++ni) {
            const __bf16* bp = &Bs[wn * 32 + ni * 16 + row16][0];
            v8bf lo = *(const v8bf*)(bp + half * 8);
            v8bf hi = *(const v8bf*)(bp + 16 + half * 8);
            bfr[ni] = __builtin_shufflevector(lo, hi, 0, 1, 2, 3, 4, 5, 6, 7,
                                              8, 9, 10, 11, 12, 13, 14, 15);
        }

        // ---- 8 WMMAs per wave per K-step
#pragma unroll
        for (int mi = 0; mi < 4; ++mi)
#pragma unroll
            for (int ni = 0; ni < 2; ++ni)
                acc[mi][ni] = __builtin_amdgcn_wmma_f32_16x16x32_bf16(
                    false, afr[mi], false, bfr[ni],
                    (short)0, acc[mi][ni], false, false);

        __syncthreads();
    }

    // ---- fused epilogue: d2 -> dist -> mask -> store + column partial sums
    float csum[2] = {0.0f, 0.0f};
#pragma unroll
    for (int mi = 0; mi < 4; ++mi) {
#pragma unroll
        for (int ni = 0; ni < 2; ++ni) {
            const int gn_l = wn * 32 + ni * 16 + row16;
            const int gn   = gnBase + gn_l;
            const float c2v = c2s[gn_l];
#pragma unroll
            for (int i = 0; i < 8; ++i) {
                const int gm_l = wm * 64 + mi * 16 + i + 8 * half;
                const int gm   = gmBase + gm_l;
                float d2 = x2s[gm_l] + c2v - 2.0f * acc[mi][ni][i];
                d2 = fmaxf(d2, 1e-12f);
                float val = sqrtf(d2) * msks[gm_l];
                nodeOut[(size_t)gm * K_DIM + gn] = val;
                csum[ni] += val;
            }
        }
    }
#pragma unroll
    for (int ni = 0; ni < 2; ++ni) {
        float v = csum[ni];
        v += __shfl_xor(v, 16); // fold the two half-wave row groups
        if (half == 0)
            atomicAdd(&accg[gnBase + wn * 32 + ni * 16 + row16], v);
    }
}

// ---------------------------------------------------------------------------
// Reduce sum(mask), then graph[k] = acc[k] / sum(mask).
__global__ void cd_finalize(const float* __restrict__ mask,
                            const float* __restrict__ acc,
                            float* __restrict__ graph) {
    __shared__ float red[1024];
    int tid = threadIdx.x;
    float s = 0.0f;
    for (int i = tid; i < N_DIM; i += 1024) s += mask[i];
    red[tid] = s;
    __syncthreads();
    for (int off = 512; off > 0; off >>= 1) {
        if (tid < off) red[tid] += red[tid + off];
        __syncthreads();
    }
    float inv = 1.0f / red[0];
    for (int k = tid; k < K_DIM; k += 1024) graph[k] = acc[k] * inv;
}

// ---------------------------------------------------------------------------
extern "C" void kernel_launch(void* const* d_in, const int* in_sizes, int n_in,
                              void* d_out, int out_size, void* d_ws, size_t ws_size,
                              hipStream_t stream) {
    const float* node = (const float*)d_in[0]; // [N, D]
    const float* mask = (const float*)d_in[1]; // [N, 1]
    const float* cent = (const float*)d_in[2]; // [K, D]
    float* out = (float*)d_out;                // [K] graph ++ [N*K] node
    float* ws  = (float*)d_ws;

    float* x2  = ws;                   // N floats
    float* c2  = ws + N_DIM;           // K floats
    float* acc = ws + N_DIM + K_DIM;   // K floats (column accumulator)

    cd_zero_acc<<<(K_DIM + 255) / 256, 256, 0, stream>>>(acc);
    cd_rownorm<<<N_DIM / 8, 256, 0, stream>>>(node, x2, N_DIM);
    cd_rownorm<<<K_DIM / 8, 256, 0, stream>>>(cent, c2, K_DIM);

    dim3 grid(N_DIM / BM, K_DIM / BN); // 256 x 16
    cd_gemm<<<grid, 256, 0, stream>>>(node, cent, mask, x2, c2, acc, out + K_DIM);

    cd_finalize<<<1, 1024, 0, stream>>>(mask, acc, out);
}
